// Model_17549236371465
// MI455X (gfx1250) — matrix operand order repaired and checked by reading, not compiled, not run on hardware
//
#include <hip/hip_runtime.h>
#include <hip/hip_bf16.h>

// MI455X / gfx1250: 2-layer LSTM forecaster.
// bf16 WMMA (v_wmma_f32_16x16x32_bf16) with fp32 accumulation.
// Recurrence: 8 WGs x 16 waves; c in VGPRs, h double-buffered in LDS (1 barrier/step),
// half of W_hh persisted in 128 VGPRs/wave, other half streamed from L2.
// gx staged in (t, b, gate) layout so each time step is a contiguous block.
// Gate activations use gfx1250 HW tanh (V_TANH_F32) when available.

typedef __attribute__((ext_vector_type(16))) __bf16 v16bf;
typedef __attribute__((ext_vector_type(8)))  __bf16 v8bf;
typedef __attribute__((ext_vector_type(8)))  float  v8f;

#define B_   128
#define T_   512
#define CIN  32
#define D_   256
#define G_   1024   // 4*D
#define DM   4
#define PRED 96
#define COUT 32

#if __has_builtin(__builtin_amdgcn_tanhf)
#define TANHF(x) __builtin_amdgcn_tanhf(x)
#else
#define TANHF(x) tanhf(x)
#endif
__device__ __forceinline__ float sigmoidf_hw(float x) {
  return fmaf(0.5f, TANHF(0.5f * x), 0.5f);   // sigma(x) = 0.5*tanh(x/2)+0.5
}

// ---- WMMA fragment loaders from row-major bf16 (row stride ld elements) ----
// A (16x32, MxK): lane&15 = M row; lane>>4 selects K sub-block {0-7,16-23} vs {8-15,24-31}.
__device__ __forceinline__ v16bf load_fragA(const __bf16* base, int ld) {
  int lane = threadIdx.x;
  const __bf16* p = base + (size_t)(lane & 15) * ld + ((lane >> 4) << 3);
  v8bf lo = *(const v8bf*)p;          // K = k0 + half*8 + 0..7
  v8bf hi = *(const v8bf*)(p + 16);   // K = k0 + 16 + half*8 + 0..7
  return __builtin_shufflevector(lo, hi, 0,1,2,3,4,5,6,7,8,9,10,11,12,13,14,15);
}

// B (32x16, KxN) from (N,K) row-major storage: lane&15 = N column,
// lane>>4 selects K sub-block {0-7,16-23} vs {8-15,24-31} (matches load_fragA).
__device__ __forceinline__ v16bf load_fragB(const __bf16* base, int ld) {
  int lane = threadIdx.x;
  const __bf16* p = base + (size_t)(lane & 15) * ld + ((lane >> 4) << 3);
  v8bf lo = *(const v8bf*)p;
  v8bf hi = *(const v8bf*)(p + 16);
  return __builtin_shufflevector(lo, hi, 0,1,2,3,4,5,6,7,8,9,10,11,12,13,14,15);
}

// ---------------- weight conversion ----------------
__global__ void f32_to_bf16_kernel(const float* __restrict__ src,
                                   __bf16* __restrict__ dst, int n) {
  int i = blockIdx.x * blockDim.x + threadIdx.x;
  if (i < n) dst[i] = (__bf16)src[i];
}

// ---------------- embedding: circular conv3 + temporal linear ----------------
__global__ void embed_kernel(const float* __restrict__ x, const float* __restrict__ xm,
                             const float* __restrict__ cw, const float* __restrict__ tw,
                             __bf16* __restrict__ out) {
  int idx = blockIdx.x * blockDim.x + threadIdx.x;   // over B*T*D
  int d = idx & (D_ - 1);
  int t = (idx >> 8) & (T_ - 1);
  int b = idx >> 17;
  int tm = (t == 0) ? (T_ - 1) : t - 1;
  int tp = (t == T_ - 1) ? 0 : t + 1;
  const float* x0 = x + ((size_t)b * T_ + tm) * CIN;
  const float* x1 = x + ((size_t)b * T_ + t ) * CIN;
  const float* x2 = x + ((size_t)b * T_ + tp) * CIN;
  const float* w  = cw + (size_t)d * CIN * 3;
  float acc = 0.f;
  #pragma unroll
  for (int c = 0; c < CIN; ++c)
    acc += x0[c] * w[c*3+0] + x1[c] * w[c*3+1] + x2[c] * w[c*3+2];
  const float* m = xm + ((size_t)b * T_ + t) * DM;
  #pragma unroll
  for (int j = 0; j < DM; ++j) acc += m[j] * tw[d * DM + j];
  out[idx] = (__bf16)acc;
}

// ---------------- gx = inp @ W_ih^T + b_ih + b_hh  (big WMMA GEMM) ----------------
// grid (BT/32, 2), block (32,8). Wave w: input rows [32*bx, +32) (2 M-tiles, B reused),
// cols n0 = by*512 + w*64 (4 N-tiles). 64 WMMA per wave.
// Output written in (t, b, gate) layout: gx[(t*B + b)*G + n].
__global__ void __launch_bounds__(256)
gx_gemm_kernel(const __bf16* __restrict__ A, const __bf16* __restrict__ Wih,
               const float* __restrict__ bih, const float* __restrict__ bhh,
               float* __restrict__ gx) {
  int lane = threadIdx.x;
  int w = threadIdx.y;
  int row0 = blockIdx.x * 32;            // row = b*T + t
  int n0 = blockIdx.y * 512 + w * 64;
  int nl = lane & 15, mhi = (lane >> 4) * 8;
  v8f acc[2][4];
  #pragma unroll
  for (int j = 0; j < 4; ++j) {
    int n = n0 + j * 16 + nl;
    float bv = bih[n] + bhh[n];
    #pragma unroll
    for (int r = 0; r < 8; ++r) { acc[0][j][r] = bv; acc[1][j][r] = bv; }
  }
  #pragma unroll
  for (int ks = 0; ks < 8; ++ks) {
    v16bf a0 = load_fragA(A + (size_t)row0 * D_ + ks * 32, D_);
    v16bf a1 = load_fragA(A + (size_t)(row0 + 16) * D_ + ks * 32, D_);
    #pragma unroll
    for (int j = 0; j < 4; ++j) {
      v16bf b = load_fragB(Wih + (size_t)(n0 + j * 16) * D_ + ks * 32, D_);
      acc[0][j] = __builtin_amdgcn_wmma_f32_16x16x32_bf16(false, a0, false, b,
                                                          (short)0, acc[0][j], false, false);
      acc[1][j] = __builtin_amdgcn_wmma_f32_16x16x32_bf16(false, a1, false, b,
                                                          (short)0, acc[1][j], false, false);
    }
  }
  int tb = row0 & (T_ - 1);              // t of first row (32 rows stay inside one b)
  int bb = row0 >> 9;                    // b
  #pragma unroll
  for (int mt = 0; mt < 2; ++mt)
    #pragma unroll
    for (int j = 0; j < 4; ++j)
      #pragma unroll
      for (int r = 0; r < 8; ++r)
        __builtin_nontemporal_store(acc[mt][j][r],
            &gx[((size_t)(tb + mt * 16 + mhi + r) * B_ + bb) * G_ + n0 + j * 16 + nl]);
}

// ---------------- LSTM recurrence ----------------
// grid 8 (16 batches each), block (32,16). Wave w owns hidden slice d0=w*16 and
// computes its i,f,g,o tiles -> gate math wave-local, c in VGPRs.
// h double-buffered in LDS -> one barrier per step.
// W_hh fragments for K=0..127 persist in 128 VGPRs; K=128..255 streamed from L2.
__global__ void __launch_bounds__(512)
lstm_rec_kernel(const float* __restrict__ gx, const __bf16* __restrict__ Whh,
                __bf16* __restrict__ hout) {
  int lane = threadIdx.x;
  int w = threadIdx.y;
  int b0 = blockIdx.x * 16;
  int d0 = w * 16;
  int nl = lane & 15, mhi = (lane >> 4) * 8;
  __shared__ __align__(16) __bf16 hl[2][16 * D_];
  for (int i = w * 32 + lane; i < 16 * D_; i += 512) hl[0][i] = (__bf16)0.0f;
  float creg[8];
  #pragma unroll
  for (int r = 0; r < 8; ++r) creg[r] = 0.f;

  // persistent half of W_hh: ks = 0..3 (K=0..127), 4 gate tiles -> 16 frags, 128 VGPRs
  v16bf bper[4][4];
  #pragma unroll
  for (int ks = 0; ks < 4; ++ks)
    #pragma unroll
    for (int q = 0; q < 4; ++q)
      bper[ks][q] = load_fragB(Whh + (size_t)(q * 256 + d0) * D_ + ks * 32, D_);

  __syncthreads();

  for (int t = 0; t < T_; ++t) {
    int rb = t & 1, wb = rb ^ 1;
    // prefetch next step's gx lines (global_prefetch_b8); gx is (t, b, g) so
    // step t+1 lives in one contiguous 512KB block.
    if (t + 1 < T_) {
      #pragma unroll
      for (int q = 0; q < 4; ++q)
        __builtin_prefetch(&gx[((size_t)(t + 1) * B_ + b0 + mhi) * G_ + q * 256 + d0 + nl], 0, 0);
    }
    v8f acc[4];
    #pragma unroll
    for (int q = 0; q < 4; ++q) {
      int n = q * 256 + d0 + nl;
      #pragma unroll
      for (int r = 0; r < 8; ++r)
        acc[q][r] = __builtin_nontemporal_load(
            &gx[((size_t)t * B_ + b0 + mhi + r) * G_ + n]);
    }
    // K = 0..127 from register-resident W_hh
    #pragma unroll
    for (int ks = 0; ks < 4; ++ks) {
      v16bf a = load_fragA(hl[rb] + ks * 32, D_);
      #pragma unroll
      for (int q = 0; q < 4; ++q)
        acc[q] = __builtin_amdgcn_wmma_f32_16x16x32_bf16(false, a, false, bper[ks][q],
                                                         (short)0, acc[q], false, false);
    }
    // K = 128..255 streamed from L2
    #pragma unroll
    for (int ks = 4; ks < 8; ++ks) {
      v16bf a = load_fragA(hl[rb] + ks * 32, D_);
      #pragma unroll
      for (int q = 0; q < 4; ++q) {
        v16bf bb = load_fragB(Whh + (size_t)(q * 256 + d0) * D_ + ks * 32, D_);
        acc[q] = __builtin_amdgcn_wmma_f32_16x16x32_bf16(false, a, false, bb,
                                                         (short)0, acc[q], false, false);
      }
    }
    float hv[8];
    #pragma unroll
    for (int r = 0; r < 8; ++r) {
      float iv = sigmoidf_hw(acc[0][r]);
      float fv = sigmoidf_hw(acc[1][r]);
      float gv = TANHF(acc[2][r]);
      float ov = sigmoidf_hw(acc[3][r]);
      creg[r] = fv * creg[r] + iv * gv;
      hv[r] = ov * TANHF(creg[r]);
    }
    #pragma unroll
    for (int r = 0; r < 8; ++r) {
      __bf16 hb = (__bf16)hv[r];
      hl[wb][(mhi + r) * D_ + d0 + nl] = hb;                               // no race: reads hit hl[rb]
      hout[((size_t)(b0 + mhi + r) * T_ + t) * D_ + d0 + nl] = hb;
    }
    __syncthreads();   // hl[wb] visible before it becomes the read buffer at t+1
  }
}

// ---------------- FC head on last PRED steps ----------------
__global__ void fc_kernel(const __bf16* __restrict__ h2, const float* __restrict__ fw,
                          const float* __restrict__ fb, float* __restrict__ out) {
  int idx = blockIdx.x * blockDim.x + threadIdx.x;   // B*PRED*COUT
  if (idx >= B_ * PRED * COUT) return;
  int o = idx & (COUT - 1);
  int tt = (idx / COUT) % PRED;
  int b = idx / (PRED * COUT);
  int t = T_ - PRED + tt;
  const __bf16* hp = h2 + ((size_t)b * T_ + t) * D_;
  const float* wp = fw + (size_t)o * D_;
  float acc = fb[o];
  #pragma unroll 8
  for (int d = 0; d < D_; ++d) acc += (float)hp[d] * wp[d];
  out[idx] = acc;
}

extern "C" void kernel_launch(void* const* d_in, const int* in_sizes, int n_in,
                              void* d_out, int out_size, void* d_ws, size_t ws_size,
                              hipStream_t stream) {
  (void)in_sizes; (void)n_in; (void)out_size; (void)ws_size;
  const float* x   = (const float*)d_in[0];
  const float* xm  = (const float*)d_in[1];
  const float* cw  = (const float*)d_in[2];
  const float* tw  = (const float*)d_in[3];
  const float* Wih = (const float*)d_in[4];
  const float* Whh = (const float*)d_in[5];
  const float* bih = (const float*)d_in[6];
  const float* bhh = (const float*)d_in[7];
  const float* fw  = (const float*)d_in[8];
  const float* fb  = (const float*)d_in[9];
  float* out = (float*)d_out;

  // workspace layout (bytes)
  char* ws = (char*)d_ws;
  const size_t NW   = (size_t)2 * G_ * D_;            // elems per weight tensor (L*4D*D)
  const size_t NACT = (size_t)B_ * T_ * D_;           // elems per activation buffer
  __bf16* wih_bf = (__bf16*)(ws);                     // 1 MB
  __bf16* whh_bf = (__bf16*)(ws + NW * 2);            // 1 MB
  __bf16* bufA   = (__bf16*)(ws + NW * 4);            // 32 MB (emb, later h2)
  __bf16* bufB   = (__bf16*)(ws + NW * 4 + NACT * 2); // 32 MB (h1)
  float*  gxbuf  = (float*) (ws + NW * 4 + NACT * 4); // 256 MB

  // weights -> bf16
  f32_to_bf16_kernel<<<(int)((NW + 255) / 256), 256, 0, stream>>>(Wih, wih_bf, (int)NW);
  f32_to_bf16_kernel<<<(int)((NW + 255) / 256), 256, 0, stream>>>(Whh, whh_bf, (int)NW);

  // embedding
  embed_kernel<<<(B_ * T_ * D_) / 256, 256, 0, stream>>>(x, xm, cw, tw, bufA);

  // layer 0
  gx_gemm_kernel<<<dim3(B_ * T_ / 32, 2), dim3(32, 8), 0, stream>>>(
      bufA, wih_bf, bih, bhh, gxbuf);
  lstm_rec_kernel<<<dim3(B_ / 16), dim3(32, 16), 0, stream>>>(gxbuf, whh_bf, bufB);

  // layer 1
  gx_gemm_kernel<<<dim3(B_ * T_ / 32, 2), dim3(32, 8), 0, stream>>>(
      bufB, wih_bf + G_ * D_, bih + G_, bhh + G_, gxbuf);
  lstm_rec_kernel<<<dim3(B_ / 16), dim3(32, 16), 0, stream>>>(
      gxbuf, whh_bf + G_ * D_, bufA);

  // FC head
  fc_kernel<<<(B_ * PRED * COUT + 255) / 256, 256, 0, stream>>>(bufA, fw, fb, out);
}
